// MemAttentionHead_83691732729962
// MI455X (gfx1250) — matrix operand order repaired and checked by reading, not compiled, not run on hardware
//
#include <hip/hip_runtime.h>
#include <stdint.h>

#define D_MODEL 1024
#define BATCH   8
#define LQ      2048
#define LK      2048
#define SCALE   0.03125f   // 1/sqrt(1024)

typedef __attribute__((ext_vector_type(16))) __bf16 v16bf;
typedef __attribute__((ext_vector_type(8)))  float  v8f;

union Frag16 {
  uint32_t u[8];
  uint4    q[2];
  v16bf    v;
};

static __device__ __forceinline__ uint32_t bf16_1(float f) {
  uint32_t u = __float_as_uint(f);
  u += 0x7FFFu + ((u >> 16) & 1u);
  return u >> 16;
}
static __device__ __forceinline__ uint32_t pack_bf16(float lo, float hi) {
  return bf16_1(lo) | (bf16_1(hi) << 16);
}
static __device__ __forceinline__ v8f zero8() {
  v8f z;
#pragma unroll
  for (int i = 0; i < 8; ++i) z[i] = 0.0f;
  return z;
}
static __device__ __forceinline__ v8f wmma_bf16(const Frag16& a, const Frag16& b, v8f c) {
  return __builtin_amdgcn_wmma_f32_16x16x32_bf16(false, a.v, false, b.v, (short)0, c,
                                                 false, false);
}

// ---------------------------------------------------------------------------
// fp32 -> bf16 elementwise convert, 8 elements / thread (b128 in, b128 out)
// ---------------------------------------------------------------------------
__global__ __launch_bounds__(256) void convert_bf16_kernel(
    const float* __restrict__ src, uint16_t* __restrict__ dst, int n8) {
  int i = blockIdx.x * 256 + threadIdx.x;
  if (i >= n8) return;
  const float4* s = (const float4*)src + (size_t)i * 2;
  float4 a = s[0];
  float4 b = s[1];
  uint4 o;
  o.x = pack_bf16(a.x, a.y);
  o.y = pack_bf16(a.z, a.w);
  o.z = pack_bf16(b.x, b.y);
  o.w = pack_bf16(b.z, b.w);
  ((uint4*)dst)[i] = o;
}

// ---------------------------------------------------------------------------
// values [B][LK][D] fp32  ->  vT [B][D][LK] bf16  (64x64 LDS tile transpose)
// ---------------------------------------------------------------------------
__global__ __launch_bounds__(256) void transpose_vals_kernel(
    const float* __restrict__ v, uint16_t* __restrict__ vt) {
  const int tilesD = D_MODEL / 64;   // 16
  const int tilesK = LK / 64;        // 32
  int t = blockIdx.x;
  const int b = t / (tilesK * tilesD);
  t %= (tilesK * tilesD);
  const int k0 = (t / tilesD) * 64;
  const int d0 = (t % tilesD) * 64;

  __shared__ __align__(16) uint16_t tile[64][72];  // padded

  const int tid = threadIdx.x;
  {
    const int r = tid >> 2;
    const int c = (tid & 3) << 4;
    const float* src = v + (size_t)(b * LK + k0 + r) * D_MODEL + d0 + c;
    float4 f0 = ((const float4*)src)[0];
    float4 f1 = ((const float4*)src)[1];
    float4 f2 = ((const float4*)src)[2];
    float4 f3 = ((const float4*)src)[3];
    tile[r][c + 0]  = (uint16_t)bf16_1(f0.x); tile[r][c + 1]  = (uint16_t)bf16_1(f0.y);
    tile[r][c + 2]  = (uint16_t)bf16_1(f0.z); tile[r][c + 3]  = (uint16_t)bf16_1(f0.w);
    tile[r][c + 4]  = (uint16_t)bf16_1(f1.x); tile[r][c + 5]  = (uint16_t)bf16_1(f1.y);
    tile[r][c + 6]  = (uint16_t)bf16_1(f1.z); tile[r][c + 7]  = (uint16_t)bf16_1(f1.w);
    tile[r][c + 8]  = (uint16_t)bf16_1(f2.x); tile[r][c + 9]  = (uint16_t)bf16_1(f2.y);
    tile[r][c + 10] = (uint16_t)bf16_1(f2.z); tile[r][c + 11] = (uint16_t)bf16_1(f2.w);
    tile[r][c + 12] = (uint16_t)bf16_1(f3.x); tile[r][c + 13] = (uint16_t)bf16_1(f3.y);
    tile[r][c + 14] = (uint16_t)bf16_1(f3.z); tile[r][c + 15] = (uint16_t)bf16_1(f3.w);
  }
  __syncthreads();
  {
    const int rd = tid >> 2;
    const int kc = (tid & 3) << 4;
    uint16_t* dst = vt + (size_t)(b * D_MODEL + d0 + rd) * LK + k0 + kc;
    uint4 o[2];
    uint16_t* tmp = (uint16_t*)o;
#pragma unroll
    for (int i = 0; i < 16; ++i) tmp[i] = tile[kc + i][rd];
    ((uint4*)dst)[0] = o[0];
    ((uint4*)dst)[1] = o[1];
  }
}

// ---------------------------------------------------------------------------
// Q projection: qout[m][e] = bf16( sum_d x[m][d]*Wq[e][d] + bq[e] )
// One wave -> 32(M) x 128(N); block = 8 waves -> 256x128.  Double-buffered.
// ---------------------------------------------------------------------------
__global__ __launch_bounds__(256) void qproj_kernel(
    const uint16_t* __restrict__ xb,   // [16384][1024] bf16
    const uint16_t* __restrict__ wb,   // [1024][1024]  bf16 (row e, col d)
    const float* __restrict__ bq,      // [1024]
    uint16_t* __restrict__ qout) {     // [16384][1024] bf16
  const int NB    = D_MODEL / 128;     // 8
  const int mBlk  = blockIdx.x / NB;
  const int nBlk  = blockIdx.x % NB;
  const int wave  = threadIdx.x >> 5;
  const int lane  = threadIdx.x & 31;
  const int lhalf = lane >> 4;
  const int l16   = lane & 15;
  const int m0 = mBlk * 256 + wave * 32;
  const int n0 = nBlk * 128;

  v8f C[16];
#pragma unroll
  for (int i = 0; i < 16; ++i) C[i] = zero8();

  const uint16_t* xrow0 = xb + (size_t)(m0 + l16) * D_MODEL + lhalf * 8;
  const uint16_t* xrow1 = xb + (size_t)(m0 + 16 + l16) * D_MODEL + lhalf * 8;
  const uint16_t* wrow  = wb + (size_t)(n0 + l16) * D_MODEL + lhalf * 8;

  Frag16 a0[2], a1[2], bf[2];
  auto loadA = [&](int kc, int buf) {
    a0[buf].q[0] = *(const uint4*)(xrow0 + kc);
    a0[buf].q[1] = *(const uint4*)(xrow0 + kc + 16);
    a1[buf].q[0] = *(const uint4*)(xrow1 + kc);
    a1[buf].q[1] = *(const uint4*)(xrow1 + kc + 16);
  };
  auto loadB = [&](int kc, int nt, int buf) {
    const uint16_t* bp = wrow + (size_t)(nt * 16) * D_MODEL + kc;
    bf[buf].q[0] = *(const uint4*)(bp);
    bf[buf].q[1] = *(const uint4*)(bp + 16);
  };

  loadA(0, 0);
  for (int kc = 0; kc < D_MODEL; kc += 32) {
    const int cur = (kc >> 5) & 1;
    loadB(kc, 0, 0);
    if (kc + 32 < D_MODEL) loadA(kc + 32, cur ^ 1);
#pragma unroll
    for (int nt = 0; nt < 8; ++nt) {
      if (nt < 7) loadB(kc, nt + 1, (nt + 1) & 1);
      C[nt]     = wmma_bf16(a0[cur], bf[nt & 1], C[nt]);
      C[8 + nt] = wmma_bf16(a1[cur], bf[nt & 1], C[8 + nt]);
    }
  }
#pragma unroll
  for (int nt = 0; nt < 8; ++nt) {
    const float bias = bq[n0 + nt * 16 + l16];
#pragma unroll
    for (int i = 0; i < 8; ++i) {
      const int r = lhalf * 8 + i;
      qout[(size_t)(m0 + r) * D_MODEL + n0 + nt * 16 + l16] =
          (uint16_t)bf16_1(C[nt][i] + bias);
      qout[(size_t)(m0 + 16 + r) * D_MODEL + n0 + nt * 16 + l16] =
          (uint16_t)bf16_1(C[8 + nt][i] + bias);
    }
  }
}

// ---------------------------------------------------------------------------
// Flash attention: block = 8 waves, 32 query rows, online softmax over LK.
// Q tile staged in LDS once per block; K / V^T fragments double-buffered.
// ---------------------------------------------------------------------------
__global__ __launch_bounds__(256) void attn_kernel(
    const uint16_t* __restrict__ qb,    // [B*LQ][D] bf16
    const uint16_t* __restrict__ kb,    // [B*LK][D] bf16
    const uint16_t* __restrict__ vtb,   // [B][D][LK] bf16
    const float* __restrict__ mask,     // [B*LK]
    float* __restrict__ out) {          // [B*LQ][D]
  const int b    = blockIdx.x / (LQ / 32);
  const int q0   = (blockIdx.x % (LQ / 32)) * 32;
  const int tid  = threadIdx.x;
  const int wave = tid >> 5;
  const int lane = tid & 31;
  const int lhalf = lane >> 4;
  const int l16   = lane & 15;

  __shared__ __align__(16) uint16_t Q_lds[32][D_MODEL];  // 64 KB
  __shared__ __align__(16) float    S_buf[32][128];      // 16 KB
  __shared__ __align__(16) uint32_t P_buf[32][64];       // 8 KB (bf16 pairs)
  __shared__ float red_max[32][8];
  __shared__ float red_sum[32][8];
  __shared__ float M_run[32], L_run[32], C_fac[32];

  v8f O[16];
#pragma unroll
  for (int i = 0; i < 16; ++i) O[i] = zero8();

  // ---- stage Q tile into LDS (once) + init running stats
  {
    const int row  = tid >> 3;
    const int col0 = (tid & 7) * 128;
    const uint4* src = (const uint4*)(qb + (size_t)(b * LQ + q0 + row) * D_MODEL + col0);
    uint4* dst = (uint4*)(&Q_lds[row][col0]);
#pragma unroll
    for (int i = 0; i < 16; ++i) dst[i] = src[i];
  }
  if (tid < 32) { M_run[tid] = -3.0e38f; L_run[tid] = 0.0f; }
  __syncthreads();

  const uint16_t* krow_base = kb + (size_t)b * LK * D_MODEL;
  const uint16_t* vt_base   = vtb + (size_t)b * D_MODEL * LK;
  const float*    mask_b    = mask + b * LK;

  for (int kt = 0; kt < LK / 128; ++kt) {
    const int keyBase = kt * 128;
    if (kt + 1 < LK / 128) {
      __builtin_prefetch(krow_base + (size_t)(keyBase + 128 + wave * 16 + l16) * D_MODEL, 0, 1);
      __builtin_prefetch(vt_base + (size_t)(wave * 128 + l16) * LK + keyBase + 128, 0, 1);
    }

    // ---- Phase 1: S = Q K^T  (this wave's 16 keys), K double-buffered
    const int myKey = keyBase + wave * 16;
    v8f S0 = zero8(), S1 = zero8();
    const uint16_t* krow = krow_base + (size_t)(myKey + l16) * D_MODEL + lhalf * 8;
    Frag16 bk[2];
    auto loadK = [&](int kc, int buf) {
      bk[buf].q[0] = *(const uint4*)(krow + kc);
      bk[buf].q[1] = *(const uint4*)(krow + kc + 16);
    };
    loadK(0, 0);
    for (int kc = 0; kc < D_MODEL; kc += 32) {
      const int cur = (kc >> 5) & 1;
      if (kc + 32 < D_MODEL) loadK(kc + 32, cur ^ 1);
      Frag16 a0, a1;
      const uint16_t* qp0 = &Q_lds[l16][kc + lhalf * 8];
      const uint16_t* qp1 = &Q_lds[16 + l16][kc + lhalf * 8];
      a0.q[0] = *(const uint4*)(qp0);
      a0.q[1] = *(const uint4*)(qp0 + 16);
      a1.q[0] = *(const uint4*)(qp1);
      a1.q[1] = *(const uint4*)(qp1 + 16);
      S0 = wmma_bf16(a0, bk[cur], S0);
      S1 = wmma_bf16(a1, bk[cur], S1);
    }
    const float mv = mask_b[myKey + l16];
#pragma unroll
    for (int i = 0; i < 8; ++i) {
      S_buf[lhalf * 8 + i][wave * 16 + l16]      = S0[i] * SCALE + mv;
      S_buf[16 + lhalf * 8 + i][wave * 16 + l16] = S1[i] * SCALE + mv;
    }
    __syncthreads();

    // ---- Phase 2: parallel online softmax (8 threads per query row)
    const int row = tid >> 3;
    const int seg = tid & 7;
    {
      float mx = -3.0e38f;
#pragma unroll
      for (int j = 0; j < 16; ++j) mx = fmaxf(mx, S_buf[row][seg * 16 + j]);
      red_max[row][seg] = mx;
    }
    __syncthreads();
    {
      const float mold = M_run[row];
      float m = mold;
#pragma unroll
      for (int s = 0; s < 8; ++s) m = fmaxf(m, red_max[row][s]);
      float lsum = 0.0f;
#pragma unroll
      for (int j = 0; j < 16; j += 2) {
        const float p0 = __expf(S_buf[row][seg * 16 + j] - m);
        const float p1 = __expf(S_buf[row][seg * 16 + j + 1] - m);
        lsum += p0 + p1;
        P_buf[row][(seg * 16 + j) >> 1] = pack_bf16(p0, p1);
      }
      red_sum[row][seg] = lsum;
    }
    __syncthreads();
    if (seg == 0) {
      const float mold = M_run[row];
      float m = mold;
      float tot = 0.0f;
#pragma unroll
      for (int s = 0; s < 8; ++s) {
        m = fmaxf(m, red_max[row][s]);
        tot += red_sum[row][s];
      }
      const float c = __expf(mold - m);
      M_run[row] = m;
      L_run[row] = L_run[row] * c + tot;
      C_fac[row] = c;
    }
    __syncthreads();

    // ---- Phase 3: rescale O, then O += P * V (this wave's 128 d-cols)
#pragma unroll
    for (int mt = 0; mt < 2; ++mt) {
      float cf[8];
#pragma unroll
      for (int i = 0; i < 8; ++i) cf[i] = C_fac[mt * 16 + lhalf * 8 + i];
#pragma unroll
      for (int nt = 0; nt < 8; ++nt) {
        v8f o = O[mt * 8 + nt];
#pragma unroll
        for (int i = 0; i < 8; ++i) o[i] *= cf[i];
        O[mt * 8 + nt] = o;
      }
    }
    const uint16_t* vrow = vt_base + (size_t)(wave * 128 + l16) * LK + keyBase + lhalf * 8;
    for (int kk = 0; kk < 128; kk += 32) {
      Frag16 a0, a1, bv[2];
      const uint16_t* pp0 = (const uint16_t*)&P_buf[l16][0] + kk + lhalf * 8;
      const uint16_t* pp1 = (const uint16_t*)&P_buf[16 + l16][0] + kk + lhalf * 8;
      a0.q[0] = *(const uint4*)(pp0);
      a0.q[1] = *(const uint4*)(pp0 + 16);
      a1.q[0] = *(const uint4*)(pp1);
      a1.q[1] = *(const uint4*)(pp1 + 16);
      auto loadV = [&](int nt, int buf) {
        const uint16_t* vp = vrow + (size_t)(nt * 16) * LK + kk;
        bv[buf].q[0] = *(const uint4*)(vp);
        bv[buf].q[1] = *(const uint4*)(vp + 16);
      };
      loadV(0, 0);
#pragma unroll
      for (int nt = 0; nt < 8; ++nt) {
        if (nt < 7) loadV(nt + 1, (nt + 1) & 1);
        O[nt]     = wmma_bf16(a0, bv[nt & 1], O[nt]);
        O[8 + nt] = wmma_bf16(a1, bv[nt & 1], O[8 + nt]);
      }
    }
    __syncthreads();
  }

  // ---- finalize: divide by softmax denominator, store fp32
  float li0[8], li1[8];
#pragma unroll
  for (int i = 0; i < 8; ++i) {
    li0[i] = 1.0f / L_run[lhalf * 8 + i];
    li1[i] = 1.0f / L_run[16 + lhalf * 8 + i];
  }
  float* outb = out + (size_t)(b * LQ + q0) * D_MODEL + wave * 128;
#pragma unroll
  for (int nt = 0; nt < 8; ++nt) {
#pragma unroll
    for (int i = 0; i < 8; ++i) {
      const int r0 = lhalf * 8 + i;
      outb[(size_t)r0 * D_MODEL + nt * 16 + l16]        = O[nt][i] * li0[i];
      outb[(size_t)(16 + r0) * D_MODEL + nt * 16 + l16] = O[8 + nt][i] * li1[i];
    }
  }
}

// ---------------------------------------------------------------------------
extern "C" void kernel_launch(void* const* d_in, const int* in_sizes, int n_in,
                              void* d_out, int out_size, void* d_ws, size_t ws_size,
                              hipStream_t stream) {
  (void)in_sizes; (void)n_in; (void)out_size; (void)ws_size;
  const float* x    = (const float*)d_in[0];
  const float* mask = (const float*)d_in[1];
  const float* keys = (const float*)d_in[2];
  const float* vals = (const float*)d_in[3];
  const float* Wq   = (const float*)d_in[4];
  const float* bq   = (const float*)d_in[5];
  float* out = (float*)d_out;

  // workspace layout (bf16 elements): q | k | vT | x | Wq  ->  ~130 MB
  const size_t NQK = (size_t)BATCH * LQ * D_MODEL;  // 16,777,216
  uint16_t* q_bf = (uint16_t*)d_ws;
  uint16_t* k_bf = q_bf + NQK;
  uint16_t* vT   = k_bf + NQK;
  uint16_t* x_bf = vT + NQK;
  uint16_t* w_bf = x_bf + NQK;

  const int n8_big = (int)(NQK / 8);                         // 2,097,152
  const int n8_w   = (int)((size_t)D_MODEL * D_MODEL / 8);   // 131,072

  convert_bf16_kernel<<<(n8_big + 255) / 256, 256, 0, stream>>>(x, x_bf, n8_big);
  convert_bf16_kernel<<<(n8_big + 255) / 256, 256, 0, stream>>>(keys, k_bf, n8_big);
  convert_bf16_kernel<<<(n8_w + 255) / 256, 256, 0, stream>>>(Wq, w_bf, n8_w);
  transpose_vals_kernel<<<BATCH * (LK / 64) * (D_MODEL / 64), 256, 0, stream>>>(vals, vT);
  qproj_kernel<<<(BATCH * LQ / 256) * (D_MODEL / 128), 256, 0, stream>>>(x_bf, w_bf, bq, q_bf);
  attn_kernel<<<BATCH * (LQ / 32), 256, 0, stream>>>(q_bf, k_bf, vT, mask, out);
}
